// DeepSpeedSelfCrossAttention_9552007266674
// MI455X (gfx1250) — compile-verified, matrix operand order checked by reading, not run on hardware
//
#include <hip/hip_runtime.h>

// ---------------------------------------------------------------------------
// DeepSpeed self-attention block for MI455X (gfx1250), bf16 WMMA pipeline.
// B=2, S=2048, HID=2048, HEADS=16, HEAD_DIM=128
// ---------------------------------------------------------------------------

#define BATCH 2
#define SEQ 2048
#define HID 2048
#define HEADS 16
#define HDIM 128
#define TOKENS (BATCH * SEQ)   // 4096
#define EPS 1e-6f
#define NEGINF (-1e9f)

typedef __bf16 bf16_t;
typedef __attribute__((ext_vector_type(16))) __bf16 v16bf;
typedef __attribute__((ext_vector_type(8)))  float  v8f;

union FragU { v16bf v; uint4 u[2]; };

#if defined(__has_builtin)
#  if __has_builtin(__builtin_amdgcn_tensor_load_to_lds)
#    define HAS_TDM 1
#  endif
#  if __has_builtin(__builtin_amdgcn_s_wait_tensorcnt)
#    define HAS_TENSORCNT 1
#  endif
#endif
#ifndef HAS_TDM
#  define HAS_TDM 0
#endif
#ifndef HAS_TENSORCNT
#  define HAS_TENSORCNT 0
#endif

__device__ __forceinline__ void wait_tensorcnt_n(int upcoming) {
#if HAS_TENSORCNT
  if (upcoming) __builtin_amdgcn_s_wait_tensorcnt(2);
  else          __builtin_amdgcn_s_wait_tensorcnt(0);
#else
  if (upcoming) asm volatile("s_wait_tensorcnt 2" ::: "memory");
  else          asm volatile("s_wait_tensorcnt 0" ::: "memory");
#endif
}

// D = A(16x32 bf16) * B(32x16 bf16) + C(16x16 f32)
__device__ __forceinline__ v8f wmma_bf16(v16bf a, v16bf b, v8f c) {
  return __builtin_amdgcn_wmma_f32_16x16x32_bf16(
      /*neg_a=*/false, a, /*neg_b=*/false, b,
      /*c_mod=*/(short)0, c, /*reuse_a=*/false, /*reuse_b=*/false);
}

// A-fragment (16x32, 16-bit) per ISA layout: lane (half,m) holds row m,
// k = [half*8 .. half*8+7] and [16+half*8 .. 16+half*8+7]  (two b128 loads).
__device__ __forceinline__ v16bf load_afrag_row(const bf16_t* rowptr, int kbase, int half) {
  FragU f;
  f.u[0] = *(const uint4*)(rowptr + kbase + half * 8);
  f.u[1] = *(const uint4*)(rowptr + kbase + 16 + half * 8);
  return f.v;
}

// B-fragment from pre-tiled memory (global or LDS): lane's 16 elems contiguous.
__device__ __forceinline__ v16bf load_bfrag(const bf16_t* base) {
  FragU f;
  f.u[0] = *(const uint4*)(base);
  f.u[1] = *(const uint4*)(base + 8);
  return f.v;
}

// inverse of the k<->(half,e) mapping for 16-bit fragments, k in [0,32)
__device__ __forceinline__ void k_to_halfe(int k, int& half, int& e) {
  half = (k >> 3) & 1;
  e = (k & 7) + (((k >> 4) & 1) << 3);
}

#if HAS_TDM
typedef unsigned int u32x4 __attribute__((ext_vector_type(4)));
typedef int i32x4 __attribute__((ext_vector_type(4)));
typedef int i32x8 __attribute__((ext_vector_type(8)));

// Generic pointer to shared memory: LDS aperture puts the LDS byte address in
// addr[31:0] (ISA 10.2 per-aperture address calculation).
__device__ __forceinline__ unsigned ldsoff(const void* p) {
  return (unsigned)(unsigned long long)(uintptr_t)p;
}

// TDM 2D tile load: t1 rows of t0_units (8-byte units), row stride
// stride0_units, contiguous into LDS at ldsaddr. D# packed per ISA 8.3/8.4.
__device__ __forceinline__ void tdm_load_tile_2d(const void* gptr, unsigned ldsaddr,
                                                 unsigned t0_units, unsigned t1,
                                                 unsigned long long stride0_units) {
  unsigned long long ga = (unsigned long long)(uintptr_t)gptr;
  u32x4 g0;
  g0.x = 1u;                                           // count=1, user descriptor
  g0.y = ldsaddr;                                      // [63:32]  lds_addr
  g0.z = (unsigned)ga;                                 // [95:64]  global_addr lo
  g0.w = (unsigned)((ga >> 32) & 0x01ffffffu) | (2u << 30);  // addr hi | type=2
  const unsigned D0 = 1u << 30;                        // tensor_dim0 (no OOB clip)
  const unsigned D1 = t1;                              // tensor_dim1
  i32x8 g1;
  g1[0] = (int)(3u << 16);                             // wg_mask=0, data_size=8B
  g1[1] = (int)((D0 & 0xffffu) << 16);                 // [63:48]   dim0 lo16
  g1[2] = (int)((D0 >> 16) | ((D1 & 0xffffu) << 16));  // dim0 hi16 | dim1 lo16
  g1[3] = (int)((D1 >> 16) | (t0_units << 16));        // dim1 hi16 | tile_dim0
  g1[4] = (int)(t1);                                   // tile_dim1, tile_dim2=0
  g1[5] = (int)(stride0_units & 0xffffffffu);          // dim0_stride lo32
  g1[6] = (int)((stride0_units >> 32) & 0xffffu);      // dim0_stride hi16
  g1[7] = 0;
  i32x4 z4 = (i32x4)0;
  i32x8 z8 = (i32x8)0;
  // 6-arg form (clang-23 / therock-10.0 headers): groups 2/3 unused (2D tile)
  __builtin_amdgcn_tensor_load_to_lds(g0, g1, z4, z4, z8, 0);
}
#endif

// ---------------------------------------------------------------------------
// Repack fp32 [K x N] row-major weights into bf16 B-fragment tiles:
// frag index = kc*(N/16)+nt, within frag: lane*16+e  (lane = half*16+n)
// ---------------------------------------------------------------------------
__global__ void convert_w_tiled(const float* __restrict__ src, bf16_t* __restrict__ dst,
                                int K, int N) {
  const int NT = N >> 4;
  const long total = (long)K * N;
  for (long t = (long)blockIdx.x * blockDim.x + threadIdx.x; t < total;
       t += (long)gridDim.x * blockDim.x) {
    int e = (int)(t & 15);
    int lane = (int)((t >> 4) & 31);
    long frag = t >> 9;
    int nt = (int)(frag % NT);
    int kc = (int)(frag / NT);
    int half = lane >> 4, n = lane & 15;
    int k32 = (e < 8) ? (half * 8 + e) : (16 + half * 8 + (e - 8));
    long srcIdx = (long)(kc * 32 + k32) * N + nt * 16 + n;
    dst[t] = (bf16_t)src[srcIdx];
  }
}

// ---------------------------------------------------------------------------
// LayerNorm: one block per token row (HID=2048, 256 threads x 8 elems)
// ---------------------------------------------------------------------------
__global__ void ln_kernel(const float* __restrict__ x, const float* __restrict__ w,
                          const float* __restrict__ b, bf16_t* __restrict__ xln) {
  __shared__ float s1[256], s2[256];
  const int row = blockIdx.x, tid = threadIdx.x;
  const float* xr = x + (long)row * HID;
  float vals[8];
  float sum = 0.f, sq = 0.f;
  for (int j = 0; j < 8; ++j) {
    float v = xr[tid + j * 256];
    vals[j] = v; sum += v; sq += v * v;
  }
  s1[tid] = sum; s2[tid] = sq;
  __syncthreads();
  for (int off = 128; off > 0; off >>= 1) {
    if (tid < off) { s1[tid] += s1[tid + off]; s2[tid] += s2[tid + off]; }
    __syncthreads();
  }
  const float mu = s1[0] * (1.f / HID);
  const float var = s2[0] * (1.f / HID) - mu * mu;
  const float rstd = rsqrtf(var + EPS);
  bf16_t* orow = xln + (long)row * HID;
  for (int j = 0; j < 8; ++j) {
    int c = tid + j * 256;
    orow[c] = (bf16_t)((vals[j] - mu) * rstd * w[c] + b[c]);
  }
}

// ---------------------------------------------------------------------------
// QKV GEMM: [4096 x 2048] @ [2048 x 6144] + bias. Ping-pong double-buffered
// K loop (register prefetch, no copy rotation), full VGPR budget via
// launch_bounds. Epilogue scatters into Q (row-major) / Kt,V (B-frag tiles).
// Block = 256 thr (8 waves), wave tile 16x64, block tile 32x256.
// ---------------------------------------------------------------------------
__global__ __launch_bounds__(256, 1) void qkv_gemm(
    const bf16_t* __restrict__ xln, const bf16_t* __restrict__ wq,
    const float* __restrict__ bias, bf16_t* __restrict__ qb,
    bf16_t* __restrict__ ktb, bf16_t* __restrict__ vb) {
  const int NT = 6144 >> 4;  // 384
  const int lane = threadIdx.x & 31, wid = threadIdx.x >> 5;
  const int wm = wid & 1, wn = wid >> 1;
  const int rowbase = blockIdx.x * 32 + wm * 16;
  const int colbase = blockIdx.y * 256 + wn * 64;
  const int half = lane >> 4, m = lane & 15, n = m;
  const int nt0 = colbase >> 4;

  v8f acc[4] = {};
  const bf16_t* arow = xln + (long)(rowbase + m) * HID;

  v16bf a0 = load_afrag_row(arow, 0, half);
  v16bf b0[4];
  for (int f = 0; f < 4; ++f)
    b0[f] = load_bfrag(wq + ((long)(nt0 + f)) * 512 + lane * 16);
  for (int kc = 0; kc < HID / 32; kc += 2) {
    v16bf a1 = load_afrag_row(arow, (kc + 1) * 32, half);
    v16bf b1[4];
    for (int f = 0; f < 4; ++f)
      b1[f] = load_bfrag(wq + ((long)(kc + 1) * NT + nt0 + f) * 512 + lane * 16);
    for (int f = 0; f < 4; ++f) acc[f] = wmma_bf16(a0, b0[f], acc[f]);
    if (kc + 2 < HID / 32) {
      a0 = load_afrag_row(arow, (kc + 2) * 32, half);
      for (int f = 0; f < 4; ++f)
        b0[f] = load_bfrag(wq + ((long)(kc + 2) * NT + nt0 + f) * 512 + lane * 16);
    }
    for (int f = 0; f < 4; ++f) acc[f] = wmma_bf16(a1, b1[f], acc[f]);
  }

  for (int f = 0; f < 4; ++f) {
    const int col = colbase + f * 16 + n;
    const int which = col >> 11;       // 0=q 1=k 2=v
    const int hc = col & 2047;
    const int h = hc >> 7, d = hc & 127;
    const float bi = bias[col];
    for (int i = 0; i < 8; ++i) {
      const int row = rowbase + half * 8 + i;
      const int bidx = row >> 11, s = row & 2047;
      const int bh = bidx * HEADS + h;
      const bf16_t bv = (bf16_t)(acc[f][i] + bi);
      if (which == 0) {
        qb[((long)bh * SEQ + s) * HDIM + d] = bv;
      } else if (which == 1) {
        int dc = d >> 5, k32 = d & 31, kh, e;
        k_to_halfe(k32, kh, e);
        int st = s >> 4, nn = s & 15;
        ktb[(((long)(bh * 4 + dc) * 128 + st) * 32 + kh * 16 + nn) * 16 + e] = bv;
      } else {
        int st = s >> 5, k32 = s & 31, kh, e;
        k_to_halfe(k32, kh, e);
        int dc = d >> 4, nn = d & 15;
        vb[(((long)(bh * 64 + st) * 8 + dc) * 32 + kh * 16 + nn) * 16 + e] = bv;
      }
    }
  }
}

// ---------------------------------------------------------------------------
// Flash attention: one wave per (b,h, 16-query tile). Online softmax,
// scores and P@V via WMMA. K/V tiles DMA'd into LDS by the Tensor Data
// Mover, double-buffered on TENSORcnt; P transposed C->A layout through LDS.
// ---------------------------------------------------------------------------
__global__ void attn_kernel(const bf16_t* __restrict__ qb, const bf16_t* __restrict__ ktb,
                            const bf16_t* __restrict__ vb, const float* __restrict__ mask,
                            bf16_t* __restrict__ ctx) {
  __shared__ __align__(16) bf16_t P[16 * 32];
#if HAS_TDM
  __shared__ __align__(16) bf16_t Kl[2][4096];  // [buf][dc(4)][frag(2)][512]
  __shared__ __align__(16) bf16_t Vl[2][4096];  // [buf][frag(8)][512]
#endif
  const int lane = threadIdx.x & 31;
  const int half = lane >> 4, m = lane & 15, n = m;
  const int qtile = blockIdx.x;          // 0..127
  const int bh = blockIdx.y;             // 0..31
  const int b = bh >> 4, h = bh & 15;
  const int qbase = qtile * 16;
  const float scale = 0.0883883476483184f; // 1/sqrt(128)

  v16bf qf[4];
  const bf16_t* qrow = qb + ((long)bh * SEQ + qbase + m) * HDIM;
  for (int dc = 0; dc < 4; ++dc) qf[dc] = load_afrag_row(qrow, dc * 32, half);

  v8f acc[8] = {};
  float mrow[8], lrow[8];
  for (int i = 0; i < 8; ++i) { mrow[i] = -3.0e38f; lrow[i] = 0.f; }

  const bf16_t* ktbh = ktb + (long)bh * 4 * 128 * 512;
  const bf16_t* vbh  = vb  + (long)bh * 64 * 8 * 512;
  const int nkt = (qbase + 16 + 31) >> 5;

#if HAS_TDM
  // prime buffer 0: K tile = 4 rows (dc) x 2KB, row stride 128KB; V tile = 8KB
  tdm_load_tile_2d(ktbh + ((long)0) * 512, ldsoff(&Kl[0][0]), 256, 4, 16384ull);
  tdm_load_tile_2d(vbh, ldsoff(&Vl[0][0]), 1024, 1, 1024ull);
#endif

  for (int kt = 0; kt < nkt; ++kt) {
    const int j0 = kt * 32;
#if HAS_TDM
    const int buf = kt & 1;
    if (kt + 1 < nkt) {
      const int nb = (kt + 1) & 1;
      tdm_load_tile_2d(ktbh + ((long)((j0 + 32) >> 4)) * 512, ldsoff(&Kl[nb][0]),
                       256, 4, 16384ull);
      tdm_load_tile_2d(vbh + (long)(kt + 1) * 8 * 512, ldsoff(&Vl[nb][0]),
                       1024, 1, 1024ull);
      wait_tensorcnt_n(1);
    } else {
      wait_tensorcnt_n(0);
    }
#endif
    v8f s0 = {}, s1 = {};
    for (int dc = 0; dc < 4; ++dc) {
#if HAS_TDM
      v16bf b0 = load_bfrag(&Kl[buf][(dc * 2 + 0) * 512 + lane * 16]);
      v16bf b1 = load_bfrag(&Kl[buf][(dc * 2 + 1) * 512 + lane * 16]);
#else
      v16bf b0 = load_bfrag(ktbh + ((long)(dc * 128 + (j0 >> 4))) * 512 + lane * 16);
      v16bf b1 = load_bfrag(ktbh + ((long)(dc * 128 + (j0 >> 4) + 1)) * 512 + lane * 16);
#endif
      s0 = wmma_bf16(qf[dc], b0, s0);
      s1 = wmma_bf16(qf[dc], b1, s1);
    }
    for (int i = 0; i < 8; ++i) {
      const int row = qbase + half * 8 + i;
      const int c0 = j0 + n, c1 = j0 + 16 + n;
      float v0 = ((c0 <= row) ? s0[i] * scale : NEGINF) + mask[b * SEQ + c0];
      float v1 = ((c1 <= row) ? s1[i] * scale : NEGINF) + mask[b * SEQ + c1];
      float rmax = fmaxf(v0, v1);
      for (int off = 8; off >= 1; off >>= 1)
        rmax = fmaxf(rmax, __shfl_xor(rmax, off, 32));
      const float mo = mrow[i];
      const float mn = fmaxf(mo, rmax);
      const float sf = __expf(mo - mn);
      const float e0 = __expf(v0 - mn);
      const float e1 = __expf(v1 - mn);
      float rs = e0 + e1;
      for (int off = 8; off >= 1; off >>= 1)
        rs += __shfl_xor(rs, off, 32);
      lrow[i] = lrow[i] * sf + rs;
      mrow[i] = mn;
      for (int dc = 0; dc < 8; ++dc) acc[dc][i] *= sf;
      const int rl = half * 8 + i;
      P[rl * 32 + n] = (bf16_t)e0;
      P[rl * 32 + 16 + n] = (bf16_t)e1;
    }
    asm volatile("s_wait_dscnt 0" ::: "memory");  // P stores visible to A-layout reads
    FragU pf;
    pf.u[0] = *(const uint4*)(P + m * 32 + half * 8);
    pf.u[1] = *(const uint4*)(P + m * 32 + 16 + half * 8);
    for (int dc = 0; dc < 8; ++dc) {
#if HAS_TDM
      v16bf vf = load_bfrag(&Vl[buf][dc * 512 + lane * 16]);
#else
      v16bf vf = load_bfrag(vbh + ((long)(kt * 8 + dc)) * 512 + lane * 16);
#endif
      acc[dc] = wmma_bf16(pf.v, vf, acc[dc]);
    }
  }

  for (int dc = 0; dc < 8; ++dc) {
    for (int i = 0; i < 8; ++i) {
      const int row = qbase + half * 8 + i;
      const long token = (long)b * SEQ + row;
      const int d = dc * 16 + n;
      ctx[token * HID + h * HDIM + d] = (bf16_t)(acc[dc][i] / lrow[i]);
    }
  }
}

// ---------------------------------------------------------------------------
// Output GEMM: ctx[4096x2048]bf16 @ Wo[2048x2048]bf16 -> out fp32
// Ping-pong double-buffered K loop.
// ---------------------------------------------------------------------------
__global__ __launch_bounds__(256, 1) void out_gemm(
    const bf16_t* __restrict__ ctx, const bf16_t* __restrict__ wo,
    float* __restrict__ out) {
  const int NT = HID >> 4;  // 128
  const int lane = threadIdx.x & 31, wid = threadIdx.x >> 5;
  const int wm = wid & 1, wn = wid >> 1;
  const int rowbase = blockIdx.x * 32 + wm * 16;
  const int colbase = blockIdx.y * 256 + wn * 64;
  const int half = lane >> 4, m = lane & 15, n = m;
  const int nt0 = colbase >> 4;

  v8f acc[4] = {};
  const bf16_t* arow = ctx + (long)(rowbase + m) * HID;

  v16bf a0 = load_afrag_row(arow, 0, half);
  v16bf b0[4];
  for (int f = 0; f < 4; ++f)
    b0[f] = load_bfrag(wo + ((long)(nt0 + f)) * 512 + lane * 16);
  for (int kc = 0; kc < HID / 32; kc += 2) {
    v16bf a1 = load_afrag_row(arow, (kc + 1) * 32, half);
    v16bf b1[4];
    for (int f = 0; f < 4; ++f)
      b1[f] = load_bfrag(wo + ((long)(kc + 1) * NT + nt0 + f) * 512 + lane * 16);
    for (int f = 0; f < 4; ++f) acc[f] = wmma_bf16(a0, b0[f], acc[f]);
    if (kc + 2 < HID / 32) {
      a0 = load_afrag_row(arow, (kc + 2) * 32, half);
      for (int f = 0; f < 4; ++f)
        b0[f] = load_bfrag(wo + ((long)(kc + 2) * NT + nt0 + f) * 512 + lane * 16);
    }
    for (int f = 0; f < 4; ++f) acc[f] = wmma_bf16(a1, b1[f], acc[f]);
  }

  for (int f = 0; f < 4; ++f) {
    const int col = colbase + f * 16 + n;
    for (int i = 0; i < 8; ++i) {
      const int row = rowbase + half * 8 + i;
      out[(long)row * HID + col] = acc[f][i];
    }
  }
}

// ---------------------------------------------------------------------------
extern "C" void kernel_launch(void* const* d_in, const int* in_sizes, int n_in,
                              void* d_out, int out_size, void* d_ws, size_t ws_size,
                              hipStream_t stream) {
  const float* input  = (const float*)d_in[0];
  const float* mask   = (const float*)d_in[1];
  const float* norm_w = (const float*)d_in[2];
  const float* norm_b = (const float*)d_in[3];
  const float* qkvw   = (const float*)d_in[4];
  const float* qkvb   = (const float*)d_in[5];
  const float* ow     = (const float*)d_in[6];
  float* out = (float*)d_out;

  char* ws = (char*)d_ws;
  size_t off = 0;
  bf16_t* Xln   = (bf16_t*)(ws + off); off += (size_t)TOKENS * HID * 2;       // 16 MB
  bf16_t* WqkvT = (bf16_t*)(ws + off); off += (size_t)HID * 3 * HID * 2;      // 24 MB
  bf16_t* WoT   = (bf16_t*)(ws + off); off += (size_t)HID * HID * 2;          //  8 MB
  bf16_t* Qb    = (bf16_t*)(ws + off); off += (size_t)TOKENS * HID * 2;       // 16 MB
  bf16_t* Ktb   = (bf16_t*)(ws + off); off += (size_t)TOKENS * HID * 2;       // 16 MB
  bf16_t* Vb    = (bf16_t*)(ws + off); off += (size_t)TOKENS * HID * 2;       // 16 MB
  bf16_t* Ctx   = (bf16_t*)(ws + off); off += (size_t)TOKENS * HID * 2;       // 16 MB
  (void)ws_size; (void)in_sizes; (void)n_in; (void)out_size;

  convert_w_tiled<<<4096, 256, 0, stream>>>(qkvw, WqkvT, HID, 3 * HID);
  convert_w_tiled<<<2048, 256, 0, stream>>>(ow, WoT, HID, HID);
  ln_kernel<<<TOKENS, 256, 0, stream>>>(input, norm_w, norm_b, Xln);
  qkv_gemm<<<dim3(TOKENS / 32, (3 * HID) / 256), 256, 0, stream>>>(Xln, WqkvT, qkvb, Qb, Ktb, Vb);
  attn_kernel<<<dim3(SEQ / 16, BATCH * HEADS), 32, 0, stream>>>(Qb, Ktb, Vb, mask, Ctx);
  out_gemm<<<dim3(TOKENS / 32, HID / 256), 256, 0, stream>>>(Ctx, WoT, out);
}
